// SimpleDH_SFNN_86474871538257
// MI455X (gfx1250) — compile-verified, matrix operand order checked by reading, not run on hardware
//
#include <hip/hip_runtime.h>
#include <hip/hip_bf16.h>
#include <stdint.h>

// ---------------------------------------------------------------------------
// Problem constants (DH-SNN reference): B=256, T=250, I=700, H=1024, BR=4, O=35
// ---------------------------------------------------------------------------
#define BSZ   256
#define TLEN  250
#define IIN   700
#define HH    1024
#define BR    4
#define OO    35

#define MROWS (BSZ * TLEN)        // 64000  rows of the big GEMM
#define NCOLS (HH * BR)           // 4096   cols of the big GEMM
#define KPAD  704                 // 700 padded to a multiple of 32
#define OPAD  48                  // 35 padded to a multiple of 16
#define KOUT  1024                // readout GEMM K (= H)

typedef __attribute__((ext_vector_type(16))) _Float16 v16h;
typedef __attribute__((ext_vector_type(8)))  _Float16 v8h;
typedef __attribute__((ext_vector_type(4)))  _Float16 v4h;
typedef __attribute__((ext_vector_type(8)))  float    v8f;

__device__ __forceinline__ float sigm(float x) { return 1.0f / (1.0f + __expf(-x)); }

// Async global->LDS copy, 16B per lane.  LDS dest address (byte offset, low 32
// bits of the flat LDS aperture address) in a VGPR, 64-bit global address in a
// VGPR pair.  Tracked by ASYNCcnt (CDNA5 ISA).
__device__ __forceinline__ void async_ld_b128(uint32_t ldsAddr, const void* gaddr) {
    asm volatile("global_load_async_to_lds_b128 %0, %1, off"
                 :: "v"(ldsAddr), "v"(gaddr) : "memory");
}
__device__ __forceinline__ void wait_async0() {
    asm volatile("s_wait_asynccnt 0x0" ::: "memory");
}

// ---------------------------------------------------------------------------
// fp32 -> f16 conversion with zero padding.
// ---------------------------------------------------------------------------
__global__ void cvt_pad_f16(const float* __restrict__ src, _Float16* __restrict__ dst,
                            int rows, int cols, int colsPad, long total) {
    long i = (long)blockIdx.x * blockDim.x + threadIdx.x;
    if (i >= total) return;
    long r = i / colsPad;
    int  c = (int)(i - r * colsPad);
    float v = (r < rows && c < cols) ? src[r * (long)cols + c] : 0.0f;
    dst[i] = (_Float16)v;
}

// ---------------------------------------------------------------------------
// GEMM1: kh[m,n] = sum_k xh[m,k] * wh[n,k] + b_dense[n]        (f16 out)
//   M = 64000, N = 4096, K = 704.   Block tile 128(M) x 128(N), 8 waves laid
//   out 4(M) x 2(N); each wave computes 32(M) x 64(N): 2 A-frags x 4 B-frags
//   = 8 v_wmma_f32_16x16x32_f16 per 32-wide K step, with each B fragment
//   feeding two WMMAs (halved LDS traffic vs 16x128 strips).
//   B tile staged via GLOBAL_LOAD_ASYNC_TO_LDS_B128 into a double-buffered
//   LDS tile (2 x 8 KB); one barrier per K iteration; s_wait_asynccnt drains
//   the async channel before publishing the staged buffer.
// ---------------------------------------------------------------------------
__global__ __launch_bounds__(256) void gemm1_wmma(
    const _Float16* __restrict__ xh, const _Float16* __restrict__ wh,
    const float* __restrict__ b_dense, _Float16* __restrict__ kh) {

    __shared__ __align__(16) _Float16 ldsB[2 * 128 * 32];   // 16 KB, double buffer

    const int  tid    = threadIdx.x;
    const int  wave   = tid >> 5;
    const int  lane   = tid & 31;
    const int  lane16 = lane & 15;
    const int  hi     = lane >> 4;            // 0: K {0-7,16-23}, 1: K {8-15,24-31}
    const int  wm     = wave & 3;             // M sub-tile (4 x 32 rows)
    const int  wn     = wave >> 2;            // N sub-tile (2 x 64 cols)
    const long M0     = (long)blockIdx.x * 128;
    const long N0     = (long)blockIdx.y * 128;

    // staging map: 128 rows x 32 halves = 512 x 16B chunks, 2 per thread
    const int sRow0 = tid >> 2;               // chunk q = tid
    const int sOff0 = (tid & 3) * 8;
    const int sRow1 = (tid + 256) >> 2;       // chunk q = tid + 256
    const int sOff1 = ((tid + 256) & 3) * 8;

    v8f acc0[4], acc1[4];
    const v8f vzero = {0.f, 0.f, 0.f, 0.f, 0.f, 0.f, 0.f, 0.f};
#pragma unroll
    for (int i = 0; i < 4; ++i) { acc0[i] = vzero; acc1[i] = vzero; }

    const _Float16* aRow0 = xh + (M0 + wm * 32 + lane16) * KPAD + hi * 8;
    const _Float16* aRow1 = aRow0 + 16 * (long)KPAD;

    const int nk = KPAD / 32;                 // 22 K steps

    // prologue: stage K-tile 0 into buffer 0
    {
        async_ld_b128((uint32_t)(uintptr_t)&ldsB[sRow0 * 32 + sOff0],
                      wh + (N0 + sRow0) * KPAD + sOff0);
        async_ld_b128((uint32_t)(uintptr_t)&ldsB[sRow1 * 32 + sOff1],
                      wh + (N0 + sRow1) * KPAD + sOff1);
        wait_async0();
    }
    __syncthreads();

    for (int kk = 0; kk < nk; ++kk) {
        const int cur = kk & 1;
        const int k0  = kk * 32;

        // kick off async staging of the next K-tile into the other buffer
        if (kk + 1 < nk) {
            const int kn = k0 + 32;
            async_ld_b128((uint32_t)(uintptr_t)&ldsB[(cur ^ 1) * 4096 + sRow0 * 32 + sOff0],
                          wh + (N0 + sRow0) * KPAD + kn + sOff0);
            async_ld_b128((uint32_t)(uintptr_t)&ldsB[(cur ^ 1) * 4096 + sRow1 * 32 + sOff1],
                          wh + (N0 + sRow1) * KPAD + kn + sOff1);
        }

        // A fragments (direct global, K-contiguous 2 x 16B chunks per lane)
        v8h alo0 = *(const v8h*)(aRow0 + k0);
        v8h ahi0 = *(const v8h*)(aRow0 + k0 + 16);
        v8h alo1 = *(const v8h*)(aRow1 + k0);
        v8h ahi1 = *(const v8h*)(aRow1 + k0 + 16);
        v16h af0, af1;
#pragma unroll
        for (int i = 0; i < 8; ++i) {
            af0[i] = alo0[i]; af0[8 + i] = ahi0[i];
            af1[i] = alo1[i]; af1[8 + i] = ahi1[i];
        }
        if (kk + 1 < nk) __builtin_prefetch(aRow0 + k0 + 32, 0, 1);

#pragma unroll
        for (int nf = 0; nf < 4; ++nf) {
            const _Float16* bp =
                &ldsB[cur * 4096 + (wn * 64 + nf * 16 + lane16) * 32 + hi * 8];
            v8h blo = *(const v8h*)bp;
            v8h bhi = *(const v8h*)(bp + 16);
            v16h bf;
#pragma unroll
            for (int i = 0; i < 8; ++i) { bf[i] = blo[i]; bf[8 + i] = bhi[i]; }
            acc0[nf] = __builtin_amdgcn_wmma_f32_16x16x32_f16(
                false, af0, false, bf, (short)0, acc0[nf], false, false);
            acc1[nf] = __builtin_amdgcn_wmma_f32_16x16x32_f16(
                false, af1, false, bf, (short)0, acc1[nf], false, false);
        }

        // drain this wave's async writes, then publish buffers to the block
        if (kk + 1 < nk) wait_async0();
        __syncthreads();
    }

    // epilogue: bias + f16 store.  C layout: c[v] at row +v+hi*8, col lane16
#pragma unroll
    for (int nf = 0; nf < 4; ++nf) {
        long  n    = N0 + wn * 64 + nf * 16 + lane16;
        float bias = b_dense[n];
#pragma unroll
        for (int v = 0; v < 8; ++v) {
            long m0 = M0 + wm * 32 + v + hi * 8;
            kh[m0 * NCOLS + n]        = (_Float16)(acc0[nf][v] + bias);
            kh[(m0 + 16) * NCOLS + n] = (_Float16)(acc1[nf][v] + bias);
        }
    }
}

// ---------------------------------------------------------------------------
// Temporal scan: per (b,h) thread, 250 sequential steps.
// ---------------------------------------------------------------------------
__global__ __launch_bounds__(256) void scan_kernel(
    const _Float16* __restrict__ kh, const float* __restrict__ tau_m,
    const float* __restrict__ tau_n, _Float16* __restrict__ spikes) {

    int idx = blockIdx.x * blockDim.x + threadIdx.x;   // < B*H = 262144
    int b = idx >> 10;
    int h = idx & 1023;

    float alpha = sigm(tau_m[h]);
    float beta0 = sigm(tau_n[h * 4 + 0]);
    float beta1 = sigm(tau_n[h * 4 + 1]);
    float beta2 = sigm(tau_n[h * 4 + 2]);
    float beta3 = sigm(tau_n[h * 4 + 3]);

    float d0 = 0.f, d1 = 0.f, d2 = 0.f, d3 = 0.f, mem = 0.f, spk = 0.f;

    const _Float16* kp = kh + ((long)b * TLEN) * NCOLS + h * 4;
    _Float16*       sp = spikes + (long)b * HH + h;

    for (int t = 0; t < TLEN; ++t) {
        v4h kv = *(const v4h*)kp;
        kp += NCOLS;
        d0 = beta0 * d0 + (1.f - beta0) * (float)kv[0];
        d1 = beta1 * d1 + (1.f - beta1) * (float)kv[1];
        d2 = beta2 * d2 + (1.f - beta2) * (float)kv[2];
        d3 = beta3 * d3 + (1.f - beta3) * (float)kv[3];
        float l = d0 + d1 + d2 + d3;
        mem = alpha * mem + (1.f - alpha) * l - 0.5f * spk;
        spk = (mem > 0.5f) ? 1.0f : 0.0f;
        sp[(long)t * BSZ * HH] = (_Float16)spk;
    }
}

// ---------------------------------------------------------------------------
// GEMM2 (readout): rs[m, o] = sum_k spikes[m,k] * woh[o,k]
//   M = 64000, N = 48 (35 padded), K = 1024.  8 waves x 16 rows, 3 N-frags.
// ---------------------------------------------------------------------------
__global__ __launch_bounds__(256) void gemm2_wmma(
    const _Float16* __restrict__ sp, const _Float16* __restrict__ woh,
    float* __restrict__ rs) {

    const int  tid    = threadIdx.x;
    const int  wave   = tid >> 5;
    const int  lane   = tid & 31;
    const int  lane16 = lane & 15;
    const int  hi     = lane >> 4;
    const long M0     = (long)blockIdx.x * 128 + wave * 16;

    v8f acc[3];
    const v8f vzero = {0.f, 0.f, 0.f, 0.f, 0.f, 0.f, 0.f, 0.f};
#pragma unroll
    for (int i = 0; i < 3; ++i) acc[i] = vzero;

    const _Float16* aRow = sp + (M0 + lane16) * KOUT + hi * 8;

    for (int k0 = 0; k0 < KOUT; k0 += 32) {
        v8h alo = *(const v8h*)(aRow + k0);
        v8h ahi = *(const v8h*)(aRow + k0 + 16);
        v16h afrag;
#pragma unroll
        for (int i = 0; i < 8; ++i) { afrag[i] = alo[i]; afrag[8 + i] = ahi[i]; }

#pragma unroll
        for (int nf = 0; nf < 3; ++nf) {
            const _Float16* bp = woh + (nf * 16 + lane16) * (long)KOUT + k0 + hi * 8;
            v8h blo = *(const v8h*)bp;
            v8h bhi = *(const v8h*)(bp + 16);
            v16h bfrag;
#pragma unroll
            for (int i = 0; i < 8; ++i) { bfrag[i] = blo[i]; bfrag[8 + i] = bhi[i]; }
            acc[nf] = __builtin_amdgcn_wmma_f32_16x16x32_f16(
                false, afrag, false, bfrag, (short)0, acc[nf], false, false);
        }
    }

#pragma unroll
    for (int nf = 0; nf < 3; ++nf) {
#pragma unroll
        for (int v = 0; v < 8; ++v) {
            long m = M0 + v + hi * 8;
            rs[m * OPAD + nf * 16 + lane16] = acc[nf][v];
        }
    }
}

// ---------------------------------------------------------------------------
// Readout leaky integrator + time mean: out[b,o] = mean_t mem_t
// ---------------------------------------------------------------------------
__global__ void out_kernel(const float* __restrict__ rs, const float* __restrict__ b_out,
                           const float* __restrict__ tau_m_out, float* __restrict__ out) {
    int idx = blockIdx.x * blockDim.x + threadIdx.x;
    if (idx >= BSZ * OO) return;
    int b = idx / OO;
    int o = idx - b * OO;

    float a  = sigm(tau_m_out[o]);
    float bo = b_out[o];
    float mem = 0.f, accv = 0.f;
    for (int t = 0; t < TLEN; ++t) {
        float r = rs[((long)t * BSZ + b) * OPAD + o] + bo;
        mem  = a * mem + (1.f - a) * r;
        accv += mem;
    }
    out[idx] = accv * (1.0f / (float)TLEN);
}

// ---------------------------------------------------------------------------
// Launch
// ---------------------------------------------------------------------------
extern "C" void kernel_launch(void* const* d_in, const int* in_sizes, int n_in,
                              void* d_out, int out_size, void* d_ws, size_t ws_size,
                              hipStream_t stream) {
    const float* x         = (const float*)d_in[0];   // (B,T,I)
    const float* W_dense   = (const float*)d_in[1];   // (H*BR, I)
    const float* b_dense   = (const float*)d_in[2];   // (H*BR,)
    const float* tau_m     = (const float*)d_in[3];   // (H,)
    const float* tau_n     = (const float*)d_in[4];   // (H,BR)
    const float* W_out     = (const float*)d_in[5];   // (O,H)
    const float* b_out     = (const float*)d_in[6];   // (O,)
    const float* tau_m_out = (const float*)d_in[7];   // (O,)
    float*       out       = (float*)d_out;           // (B,O)

    // workspace layout (all sizes 256B-aligned)
    char* ws = (char*)d_ws;
    _Float16* xh  = (_Float16*)ws;  ws += (size_t)MROWS * KPAD * 2;   //  90.1 MB
    _Float16* wh  = (_Float16*)ws;  ws += (size_t)NCOLS * KPAD * 2;   //   5.8 MB
    _Float16* kh  = (_Float16*)ws;  ws += (size_t)MROWS * NCOLS * 2;  // 524.3 MB
    _Float16* sph = (_Float16*)ws;  ws += (size_t)MROWS * HH * 2;     // 131.1 MB
    _Float16* woh = (_Float16*)ws;  ws += (size_t)OPAD * KOUT * 2;    //   0.1 MB
    float*    rs  = (float*)ws;     ws += (size_t)MROWS * OPAD * 4;   //  12.3 MB

    // 1) convert + pad inputs to f16
    {
        long total = (long)MROWS * KPAD;
        cvt_pad_f16<<<(unsigned)((total + 255) / 256), 256, 0, stream>>>(
            x, xh, MROWS, IIN, KPAD, total);
    }
    {
        long total = (long)NCOLS * KPAD;
        cvt_pad_f16<<<(unsigned)((total + 255) / 256), 256, 0, stream>>>(
            W_dense, wh, NCOLS, IIN, KPAD, total);
    }
    {
        long total = (long)OPAD * KOUT;
        cvt_pad_f16<<<(unsigned)((total + 255) / 256), 256, 0, stream>>>(
            W_out, woh, OO, KOUT, KOUT, total);
    }

    // 2) big WMMA GEMM: k currents (fused bias), async-LDS double buffered
    gemm1_wmma<<<dim3(MROWS / 128, NCOLS / 128), 256, 0, stream>>>(xh, wh, b_dense, kh);

    // 3) dendrite/LIF temporal scan -> spike train (f16)
    scan_kernel<<<(BSZ * HH) / 256, 256, 0, stream>>>(kh, tau_m, tau_n, sph);

    // 4) readout WMMA GEMM: spikes @ W_out^T
    gemm2_wmma<<<MROWS / 128, 256, 0, stream>>>(sph, woh, rs);

    // 5) readout leaky integrator + mean over time
    out_kernel<<<(BSZ * OO + 255) / 256, 256, 0, stream>>>(rs, b_out, tau_m_out, out);
}
